// KANLinearNoRes_38568806318402
// MI455X (gfx1250) — matrix operand order repaired and checked by reading, not compiled, not run on hardware
//
#include <hip/hip_runtime.h>

typedef __attribute__((ext_vector_type(16))) __bf16 v16bf;
typedef __attribute__((ext_vector_type(8)))  float  v8f;

#define IN_FEATURE 1024
#define OUT_FEATURE 1024
#define BATCH 4096
#define GCOEF 8                       // GRID_SIZE + SPLINE_ORDER
#define KDIM (IN_FEATURE * GCOEF)     // 8192
#define KT (KDIM / 32)                // 256 K-tiles of 32
#define NT (OUT_FEATURE / 16)         // 64 N-tiles of 16
#define MT (BATCH / 16)               // 256 M-tiles of 16

// Scheduling fence: nothing may be reordered across this point (no runtime cost).
#define SCHED_FENCE() __builtin_amdgcn_sched_barrier(0)

static __device__ __forceinline__ unsigned short f32_to_bf16(float f) {
  unsigned int u = __float_as_uint(f);
  u += 0x7FFFu + ((u >> 16) & 1u);    // round-to-nearest-even
  return (unsigned short)(u >> 16);
}

// ---------------------------------------------------------------------------
// Kernel 1: evaluate cubic B-spline bases for each (b, i) and scatter the 8
// bf16 values straight into the WMMA A-fragment layout (16-bit A 16x32):
//   lane = hs*16 + (b&15), hs = i&1 ; element offset e = g + ((i&2) ? 8 : 0)
// A_swz linear index (ushorts): ((mTile*KT + kTile)*32 + lane)*16 + e
// The 8 coefficients of one (b,i) are contiguous -> one 16-byte store.
// ---------------------------------------------------------------------------
__global__ void __launch_bounds__(256) kan_bases_kernel(
    const float* __restrict__ x, unsigned short* __restrict__ A) {
  const unsigned int gid = blockIdx.x * blockDim.x + threadIdx.x;
  const unsigned int b = gid >> 10;        // / IN_FEATURE
  const unsigned int i = gid & 1023;
  const float xv = x[(size_t)b * IN_FEATURE + i];

  const float h = 0.4f;                    // (1 - (-1)) / GRID_SIZE
  float bas[11];
#pragma unroll
  for (int j = 0; j < 11; ++j) {
    float t0 = -1.0f + (float)(j - 3) * h; // knot t_j
    bas[j] = (xv >= t0 && xv < t0 + h) ? 1.0f : 0.0f;
  }
#pragma unroll
  for (int k = 1; k <= 3; ++k) {
    const float inv = 1.0f / ((float)k * h); // uniform: t_{j+k}-t_j = k*h
#pragma unroll
    for (int j = 0; j + k < 11; ++j) {
      float tj   = -1.0f + (float)(j - 3) * h;
      float tjk1 = tj + (float)(k + 1) * h;
      bas[j] = (xv - tj) * inv * bas[j] + (tjk1 - xv) * inv * bas[j + 1];
    }
  }

  unsigned int p[4];
#pragma unroll
  for (int g = 0; g < 4; ++g)
    p[g] = (unsigned int)f32_to_bf16(bas[2 * g]) |
           ((unsigned int)f32_to_bf16(bas[2 * g + 1]) << 16);

  const unsigned int mTile = b >> 4, row = b & 15;
  const unsigned int kTile = i >> 2;
  const unsigned int hs    = i & 1;
  const unsigned int eBase = (i & 2) ? 8u : 0u;
  const unsigned int lane  = (hs << 4) | row;
  const size_t idx = (((size_t)mTile * KT + kTile) * 32 + lane) * 16 + eBase;
  uint4 v; v.x = p[0]; v.y = p[1]; v.z = p[2]; v.w = p[3];
  *reinterpret_cast<uint4*>(A + idx) = v;   // 16B aligned (eBase in {0,8})
}

// ---------------------------------------------------------------------------
// Kernel 2: gather spline_weight[o][i][g] into the WMMA B-fragment layout
// (16-bit B 32x16): lane = hs*16 + col, e = K within half; linear index:
//   ((kTile*NT + nTile)*32 + lane)*16 + e
// Consecutive threads read/write consecutive addresses -> coalesced.
// ---------------------------------------------------------------------------
__global__ void __launch_bounds__(256) kan_wswz_kernel(
    const float* __restrict__ w, unsigned short* __restrict__ W) {
  const unsigned int tid   = blockIdx.x * blockDim.x + threadIdx.x;
  const unsigned int e     = tid & 15;
  const unsigned int lane  = (tid >> 4) & 31;
  const unsigned int frag  = tid >> 9;
  const unsigned int nTile = frag & (NT - 1);
  const unsigned int kTile = frag >> 6;
  const unsigned int o = nTile * 16 + (lane & 15);
  const unsigned int k = kTile * 32 + (lane >> 4) * 8 + (e & 7) + (e >> 3) * 16;  // k = i*8 + g
  W[tid] = f32_to_bf16(w[(size_t)o * KDIM + k]);
}

// ---------------------------------------------------------------------------
// Kernel 3: C = A x W with v_wmma_f32_16x16x32_bf16.
// Each wave: 2 M-tiles x 4 N-tiles (8 independent f32 accumulators).
// Hand-pipelined ping-pong double buffer. sched_barrier fences keep the
// [compute][load][compute][load] sections in place, and the K-loop must NOT
// be unrolled by the compiler: the body is already two pipeline stages, and
// unrolling duplicates fragment sets past the 256-VGPR window (seen in
// round 4: s_set_vgpr_msb + accumulator copies). unroll(disable) pins it.
// ---------------------------------------------------------------------------
__global__ void __launch_bounds__(256) kan_gemm_kernel(
    const unsigned short* __restrict__ A, const unsigned short* __restrict__ W,
    float* __restrict__ out) {
  const int lane = threadIdx.x & 31;
  const int wave = threadIdx.x >> 5;
  const int mTile0 = blockIdx.x * 2;                 // 2 M-tiles per block/wave
  const int nTileBase = blockIdx.y * 32 + wave * 4;  // 4 N-tiles per wave

  const v16bf* a0p = reinterpret_cast<const v16bf*>(A) +
                     (size_t)(mTile0 + 0) * KT * 32 + lane;
  const v16bf* a1p = reinterpret_cast<const v16bf*>(A) +
                     (size_t)(mTile0 + 1) * KT * 32 + lane;
  const v16bf* wp  = reinterpret_cast<const v16bf*>(W) +
                     (size_t)nTileBase * 32 + lane;

  v8f acc[2][4];
#pragma unroll
  for (int m = 0; m < 2; ++m)
#pragma unroll
    for (int j = 0; j < 4; ++j) acc[m][j] = (v8f){};

  v16bf fa0[2], fa1[2], fb[2][4];    // ping-pong fragment sets

  auto loadFrags = [&](int kt, int s) {
    fa0[s] = a0p[(size_t)kt * 32];
    fa1[s] = a1p[(size_t)kt * 32];
#pragma unroll
    for (int j = 0; j < 4; ++j) fb[s][j] = wp[(size_t)kt * NT * 32 + j * 32];
  };
  auto compute = [&](int s) {
#pragma unroll
    for (int j = 0; j < 4; ++j) {
      acc[0][j] = __builtin_amdgcn_wmma_f32_16x16x32_bf16(
          false, fa0[s], false, fb[s][j], (short)0, acc[0][j], false, false);
      acc[1][j] = __builtin_amdgcn_wmma_f32_16x16x32_bf16(
          false, fa1[s], false, fb[s][j], (short)0, acc[1][j], false, false);
    }
  };

  loadFrags(0, 0);
  loadFrags(1, 1);
#pragma clang loop unroll(disable)
  for (int kt = 0; kt + 2 < KT; kt += 2) {
    SCHED_FENCE();
    compute(0);                    // consume slot 0 (step kt)
    SCHED_FENCE();
    loadFrags(kt + 2, 0);          // refill slot 0 for step kt+2
    SCHED_FENCE();
    compute(1);                    // consume slot 1 (step kt+1)
    SCHED_FENCE();
    loadFrags(kt + 3, 1);          // refill slot 1 for step kt+3 (always < KT)
  }
  SCHED_FENCE();
  compute(0);                      // kt = KT-2
  compute(1);                      // kt = KT-1

  // C/D layout: VGPR r, lanes 0-15 -> row r, lanes 16-31 -> row 8+r; col=lane&15
  const int rowInTile = (lane >> 4) * 8;
  const int col       = lane & 15;
#pragma unroll
  for (int m = 0; m < 2; ++m) {
    float* o0 = out +
        (size_t)((mTile0 + m) * 16 + rowInTile) * OUT_FEATURE +
        nTileBase * 16 + col;
#pragma unroll
    for (int r = 0; r < 8; ++r) {
      o0[(size_t)r * OUT_FEATURE +  0] = acc[m][0][r];
      o0[(size_t)r * OUT_FEATURE + 16] = acc[m][1][r];
      o0[(size_t)r * OUT_FEATURE + 32] = acc[m][2][r];
      o0[(size_t)r * OUT_FEATURE + 48] = acc[m][3][r];
    }
  }
}

extern "C" void kernel_launch(void* const* d_in, const int* in_sizes, int n_in,
                              void* d_out, int out_size, void* d_ws, size_t ws_size,
                              hipStream_t stream) {
  const float* x  = (const float*)d_in[0];   // (4096, 1024) f32
  const float* sw = (const float*)d_in[1];   // (1024, 1024, 8) f32
  float* out = (float*)d_out;                // (4096, 1024) f32

  unsigned short* A = (unsigned short*)d_ws;            // 64 MB swizzled bases
  unsigned short* W = A + (size_t)BATCH * KDIM;         // 16 MB swizzled weights

  kan_bases_kernel<<<(BATCH * IN_FEATURE) / 256, 256, 0, stream>>>(x, A);
  kan_wswz_kernel<<<(KDIM * OUT_FEATURE) / 256, 256, 0, stream>>>(sw, W);
  kan_gemm_kernel<<<dim3(MT / 2, 2, 1), 256, 0, stream>>>(A, W, out);
}
